// PyFlashAttention_15951508537570
// MI455X (gfx1250) — compile-verified, matrix operand order checked by reading, not run on hardware
//
#include <hip/hip_runtime.h>

// ---------------------------------------------------------------------------
// Causal grouped-query flash attention for MI455X (gfx1250, wave32, WMMA).
// B=2, HQ=32, HKV=8, S=2048, D=128, fp32 in/out; f16 WMMA math, fp32 softmax.
// ---------------------------------------------------------------------------

typedef _Float16 v16h __attribute__((ext_vector_type(16)));
typedef _Float16 v8h  __attribute__((ext_vector_type(8)));
typedef float    v8f  __attribute__((ext_vector_type(8)));
typedef float    v4f  __attribute__((ext_vector_type(4)));

#define HQ_    32
#define HKV_   8
#define SEQ_   2048
#define DH_    128
#define GROUP_ (HQ_ / HKV_)
#define QTILE_ 128
#define KTILE_ 32
#define NEG_BIG (-3.0e38f)

// Build a 16x32 f16 A/B fragment from an LDS row (row stride arbitrary,
// caller passes row base). Per ISA 7.12.2: lane holds halves at K offsets
// [hi*8 .. hi*8+7] and [16+hi*8 .. 16+hi*8+7]  (hi = lane>>4).
__device__ __forceinline__ v16h frag_from_lds(const _Float16* rowbase, int hi) {
    const v8h lo = *(const v8h*)(rowbase + hi * 8);
    const v8h hc = *(const v8h*)(rowbase + 16 + hi * 8);
    return __builtin_shufflevector(lo, hc, 0,1,2,3,4,5,6,7,8,9,10,11,12,13,14,15);
}

// Same fragment layout but sourced from global fp32 with conversion.
__device__ __forceinline__ v16h frag_from_global_f32(const float* rowbase, int hi) {
    const v4f f0 = *(const v4f*)(rowbase + hi * 8);
    const v4f f1 = *(const v4f*)(rowbase + hi * 8 + 4);
    const v4f f2 = *(const v4f*)(rowbase + 16 + hi * 8);
    const v4f f3 = *(const v4f*)(rowbase + 16 + hi * 8 + 4);
    v16h a;
#pragma unroll
    for (int i = 0; i < 4; ++i) {
        a[i]      = (_Float16)f0[i];
        a[4 + i]  = (_Float16)f1[i];
        a[8 + i]  = (_Float16)f2[i];
        a[12 + i] = (_Float16)f3[i];
    }
    return a;
}

__global__ __launch_bounds__(256)
void fa_fwd_gfx1250(const float* __restrict__ q, const float* __restrict__ k,
                    const float* __restrict__ v, float* __restrict__ out) {
    __shared__ alignas(16) _Float16 lds_k[KTILE_ * DH_];    // [key][dim]   8 KB
    __shared__ alignas(16) _Float16 lds_v[DH_ * KTILE_];    // [dim][key]   8 KB (transposed V)
    __shared__ alignas(16) _Float16 lds_p[8][16 * KTILE_];  // per-wave P   8 KB

    const int tid  = threadIdx.x;
    const int lane = tid & 31;
    const int wv   = tid >> 5;
    const int col  = lane & 15;   // N / M index within a 16-wide tile
    const int hi   = lane >> 4;   // which half of the wave

    const int b    = blockIdx.z;
    const int h    = blockIdx.y;
    const int hkv  = h / GROUP_;
    const int q0   = blockIdx.x * QTILE_;
    const int qbase = q0 + wv * 16;       // this wave's 16 query rows

    const float scale = 0.08838834764831845f;  // 1/sqrt(128)

    const float* qhead = q + ((size_t)(b * HQ_ + h) * SEQ_) * DH_;
    const float* khead = k + ((size_t)(b * HKV_ + hkv) * SEQ_) * DH_;
    const float* vhead = v + ((size_t)(b * HKV_ + hkv) * SEQ_) * DH_;
    float*       ohead = out + ((size_t)(b * HQ_ + h) * SEQ_) * DH_;

    // Q fragments: 16 rows x 128 dims -> 4 chunks of 16x32 f16
    v16h qf[4];
    {
        const float* qrow = qhead + (size_t)(qbase + col) * DH_;
#pragma unroll
        for (int c = 0; c < 4; ++c) qf[c] = frag_from_global_f32(qrow + 32 * c, hi);
    }

    v8f acc[8];
#pragma unroll
    for (int t = 0; t < 8; ++t)
#pragma unroll
        for (int r = 0; r < 8; ++r) acc[t][r] = 0.0f;

    float mrow[8], lrow[8];
#pragma unroll
    for (int r = 0; r < 8; ++r) { mrow[r] = NEG_BIG; lrow[r] = 0.0f; }

    const int nblocks = q0 / KTILE_ + (QTILE_ / KTILE_);  // last key needed: q0+127

    for (int kb = 0; kb < nblocks; ++kb) {
        const int kstart = kb * KTILE_;

        // ---- cooperative staging: K (row-major) and V (transposed) -> LDS f16 ----
        {
            const int key = tid >> 3;            // 0..31
            const int seg = (tid & 7) * 16;      // 0..112
            const float* krow = khead + (size_t)(kstart + key) * DH_ + seg;
            const float* vrow = vhead + (size_t)(kstart + key) * DH_ + seg;
            const v4f kf0 = *(const v4f*)(krow + 0),  kf1 = *(const v4f*)(krow + 4);
            const v4f kf2 = *(const v4f*)(krow + 8),  kf3 = *(const v4f*)(krow + 12);
            const v4f vf0 = *(const v4f*)(vrow + 0),  vf1 = *(const v4f*)(vrow + 4);
            const v4f vf2 = *(const v4f*)(vrow + 8),  vf3 = *(const v4f*)(vrow + 12);
            _Float16* kd = &lds_k[key * DH_ + seg];
#pragma unroll
            for (int i = 0; i < 4; ++i) {
                kd[i]      = (_Float16)kf0[i];
                kd[4 + i]  = (_Float16)kf1[i];
                kd[8 + i]  = (_Float16)kf2[i];
                kd[12 + i] = (_Float16)kf3[i];
            }
#pragma unroll
            for (int i = 0; i < 4; ++i) {
                lds_v[(seg + i)      * KTILE_ + key] = (_Float16)vf0[i];
                lds_v[(seg + 4 + i)  * KTILE_ + key] = (_Float16)vf1[i];
                lds_v[(seg + 8 + i)  * KTILE_ + key] = (_Float16)vf2[i];
                lds_v[(seg + 12 + i) * KTILE_ + key] = (_Float16)vf3[i];
            }
            if (kb + 1 < nblocks) {  // hint next tile into cache (global_prefetch_b8)
                __builtin_prefetch(krow + KTILE_ * DH_, 0, 1);
                __builtin_prefetch(vrow + KTILE_ * DH_, 0, 1);
            }
        }
        __syncthreads();

        // wave-uniform: does this wave still have keys in range? (EXEC stays all-1s)
        if (kstart <= qbase + 15) {
            // ---- scores: S = Q * K^T  (two 16x16 N tiles, K-dim 128 in 4 chunks) ----
            v8f s0, s1;
#pragma unroll
            for (int r = 0; r < 8; ++r) { s0[r] = 0.0f; s1[r] = 0.0f; }
#pragma unroll
            for (int c = 0; c < 4; ++c) {
                const v16h k0 = frag_from_lds(&lds_k[(col)      * DH_ + 32 * c], hi);
                const v16h k1 = frag_from_lds(&lds_k[(16 + col) * DH_ + 32 * c], hi);
                s0 = __builtin_amdgcn_wmma_f32_16x16x32_f16(false, qf[c], false, k0,
                                                            (short)0, s0, false, false);
                s1 = __builtin_amdgcn_wmma_f32_16x16x32_f16(false, qf[c], false, k1,
                                                            (short)0, s1, false, false);
            }

            // ---- online softmax over this 32-key block ----
            const bool need_mask = (kstart + KTILE_ - 1 > qbase);
            float pr0[8], pr1[8], alpha[8];
#pragma unroll
            for (int r = 0; r < 8; ++r) {
                float a0 = s0[r] * scale;
                float a1 = s1[r] * scale;
                if (need_mask) {
                    const int qrow = qbase + r + 8 * hi;   // C-layout row for VGPR r
                    if (kstart + col > qrow)      a0 = NEG_BIG;
                    if (kstart + 16 + col > qrow) a1 = NEG_BIG;
                }
                float rm = fmaxf(a0, a1);
#pragma unroll
                for (int off = 8; off >= 1; off >>= 1)
                    rm = fmaxf(rm, __shfl_xor(rm, off, 32));
                const float mnew = fmaxf(mrow[r], rm);
                const float p0 = __expf(a0 - mnew);
                const float p1 = __expf(a1 - mnew);
                float rs = p0 + p1;
#pragma unroll
                for (int off = 8; off >= 1; off >>= 1)
                    rs += __shfl_xor(rs, off, 32);
                alpha[r] = __expf(mrow[r] - mnew);
                lrow[r]  = lrow[r] * alpha[r] + rs;
                mrow[r]  = mnew;
                pr0[r] = p0;
                pr1[r] = p1;
            }

            // ---- C-layout -> A-layout via per-wave LDS round trip ----
            _Float16* pb = &lds_p[wv][0];
#pragma unroll
            for (int r = 0; r < 8; ++r) {
                const int row = r + 8 * hi;
                pb[row * KTILE_ + col]      = (_Float16)pr0[r];
                pb[row * KTILE_ + 16 + col] = (_Float16)pr1[r];
            }
            // cross-lane within-wave dependency: wait for the DS stores to land
            __asm__ volatile("s_wait_dscnt 0" ::: "memory");
            const v16h pf = frag_from_lds(&pb[col * KTILE_], hi);

            // ---- O = alpha*O + P * V  (8 N-tiles over D=128) ----
#pragma unroll
            for (int t = 0; t < 8; ++t) {
                v8f c = acc[t];
#pragma unroll
                for (int r = 0; r < 8; ++r) c[r] *= alpha[r];
                const v16h vf = frag_from_lds(&lds_v[(t * 16 + col) * KTILE_], hi);
                acc[t] = __builtin_amdgcn_wmma_f32_16x16x32_f16(false, pf, false, vf,
                                                                (short)0, c, false, false);
            }
        }
        __syncthreads();  // protect LDS tiles before next staging pass
    }

    // ---- normalize and write out (fp32) ----
#pragma unroll
    for (int t = 0; t < 8; ++t) {
#pragma unroll
        for (int r = 0; r < 8; ++r) {
            const float o = acc[t][r] / lrow[r];
            ohead[(size_t)(qbase + r + 8 * hi) * DH_ + t * 16 + col] = o;
        }
    }
}

extern "C" void kernel_launch(void* const* d_in, const int* in_sizes, int n_in,
                              void* d_out, int out_size, void* d_ws, size_t ws_size,
                              hipStream_t stream) {
    const float* q = (const float*)d_in[0];
    const float* k = (const float*)d_in[1];
    const float* v = (const float*)d_in[2];
    float* out = (float*)d_out;

    const int B = in_sizes[0] / (HQ_ * SEQ_ * DH_);

    dim3 grid(SEQ_ / QTILE_, HQ_, B);
    dim3 block(256, 1, 1);
    fa_fwd_gfx1250<<<grid, block, 0, stream>>>(q, k, v, out);
}
